// MoELayer_15135464751558
// MI455X (gfx1250) — matrix operand order repaired and checked by reading, not compiled, not run on hardware
//
#include <hip/hip_runtime.h>
#include <hip/hip_bf16.h>

typedef __attribute__((ext_vector_type(16))) __bf16 v16bf;
typedef __attribute__((ext_vector_type(8)))  __bf16 v8bf;
typedef __attribute__((ext_vector_type(8)))  float  v8f;

#define T_TOK 8192
#define DM    1024
#define DF    4096
#define NE    8
#define MT    64          // tokens per FFN tile
#define FC    64          // d_ff chunk per iteration
#define TILES_PER_E 128   // T_TOK / MT
#define XS_STRIDE (DM + 8)
#define HS_STRIDE (FC + 8)
#define FFN_LDS_BYTES (MT*XS_STRIDE*2 + MT*HS_STRIDE*2 + MT*12)

// ---------------------------------------------------------------- utilities
__device__ __forceinline__ float gelu_exact(float v) {
    return 0.5f * v * (1.0f + erff(v * 0.70710678118654752f));
}

// ---------------------------------------------------------------- init
__global__ void moe_zero(int* __restrict__ cnt, int* __restrict__ loadcnt) {
    if (threadIdx.x < NE) { cnt[threadIdx.x] = 0; loadcnt[threadIdx.x] = 0; }
}

// ---------------------------------------------------------------- router
// One wave (32 lanes) per token. grid = T_TOK/8 blocks of 256 threads.
__global__ __launch_bounds__(256) void moe_router(
    const float* __restrict__ x, const float* __restrict__ gw,
    int* __restrict__ cnt, int* __restrict__ loadcnt,
    int* __restrict__ idxl, float* __restrict__ wgtl, int* __restrict__ sltl,
    float* __restrict__ imp_partial)
{
    __shared__ float gs[NE][DM];
    __shared__ float pb[8][NE];
    const int tid = threadIdx.x;
    for (int i = tid; i < NE * DM; i += 256) gs[i >> 10][i & (DM - 1)] = gw[i];
    __syncthreads();

    const int wid = tid >> 5, lane = tid & 31;
    const int t = blockIdx.x * 8 + wid;
    const float* xr = x + (size_t)t * DM;

    float acc[NE];
#pragma unroll
    for (int e = 0; e < NE; ++e) acc[e] = 0.f;
    for (int d = lane; d < DM; d += 32) {
        float xv = xr[d];
#pragma unroll
        for (int e = 0; e < NE; ++e) acc[e] += xv * gs[e][d];
    }
#pragma unroll
    for (int e = 0; e < NE; ++e) {
#pragma unroll
        for (int off = 16; off > 0; off >>= 1)
            acc[e] += __shfl_xor(acc[e], off, 32);
    }

    if (lane == 0) {
        float mx = acc[0];
#pragma unroll
        for (int e = 1; e < NE; ++e) mx = fmaxf(mx, acc[e]);
        float p[NE], s = 0.f;
#pragma unroll
        for (int e = 0; e < NE; ++e) { p[e] = expf(acc[e] - mx); s += p[e]; }
        float inv = 1.f / s;
#pragma unroll
        for (int e = 0; e < NE; ++e) { p[e] *= inv; pb[wid][e] = p[e]; }
        // top-2, first-occurrence tie break (matches lax.top_k)
        int e1 = 0;
#pragma unroll
        for (int e = 1; e < NE; ++e) if (p[e] > p[e1]) e1 = e;
        int e2 = (e1 == 0) ? 1 : 0;
#pragma unroll
        for (int e = 0; e < NE; ++e) if (e != e2 && e != e1 && p[e] > p[e2]) e2 = e;
        float wsum = p[e1] + p[e2];
        float g1 = p[e1] / wsum, g2 = p[e2] / wsum;
        int pos1 = atomicAdd(&cnt[e1], 1);
        idxl[e1 * T_TOK + pos1] = t; wgtl[e1 * T_TOK + pos1] = g1; sltl[e1 * T_TOK + pos1] = 0;
        atomicAdd(&loadcnt[e1], 1);
        int pos2 = atomicAdd(&cnt[e2], 1);
        idxl[e2 * T_TOK + pos2] = t; wgtl[e2 * T_TOK + pos2] = g2; sltl[e2 * T_TOK + pos2] = 1;
        atomicAdd(&loadcnt[e2], 1);
    }
    __syncthreads();
    if (tid < NE) {   // fixed-order per-block importance partial (deterministic)
        float s = 0.f;
#pragma unroll
        for (int w = 0; w < 8; ++w) s += pb[w][tid];
        imp_partial[blockIdx.x * NE + tid] = s;
    }
}

// ---------------------------------------------------------------- aux loss
__global__ void moe_aux(const float* __restrict__ imp_partial,
                        const int* __restrict__ loadcnt,
                        float* __restrict__ out_aux)
{
    __shared__ float v[NE];
    const int tid = threadIdx.x;
    if (tid < NE) {
        float s = 0.f;
        for (int b = 0; b < T_TOK / 8; ++b) s += imp_partial[b * NE + tid];
        float imp = s / (float)T_TOK;
        float ld  = (float)loadcnt[tid] / (float)(T_TOK * 2);
        v[tid] = imp * ld;
    }
    __syncthreads();
    if (tid == 0) {
        float s = 0.f;
#pragma unroll
        for (int e = 0; e < NE; ++e) s += v[e];
        *out_aux = (float)NE * s;
    }
}

// ---------------------------------------------------------------- fused expert FFN
// grid = NE * TILES_PER_E blocks, 512 threads (16 waves).
// Per block: 64 tokens of one expert; out slice per wave = 64 tokens x 64 d_model.
__global__ __launch_bounds__(512) void moe_ffn(
    const float* __restrict__ x,  const float* __restrict__ w1,
    const float* __restrict__ b1, const float* __restrict__ w2,
    const float* __restrict__ b2,
    const int* __restrict__ cnt,  const int* __restrict__ idxl,
    const float* __restrict__ wgtl, const int* __restrict__ sltl,
    float* __restrict__ planes)
{
    extern __shared__ char smem[];
    __bf16* Xs = (__bf16*)smem;                    // [MT][XS_STRIDE] bf16
    __bf16* Hs = Xs + MT * XS_STRIDE;              // [MT][HS_STRIDE] bf16
    int*   tok = (int*)(Hs + MT * HS_STRIDE);
    float* wgs = (float*)(tok + MT);
    int*   sls = (int*)(wgs + MT);

    const int e    = blockIdx.x >> 7;
    const int tile = blockIdx.x & (TILES_PER_E - 1);
    const int cnte = cnt[e];
    const int base = tile * MT;
    if (base >= cnte) return;                      // uniform across block
    const int nvalid = min(MT, cnte - base);

    const int tid = threadIdx.x;
    if (tid < MT) {
        int ok = tid < nvalid;
        int a  = e * T_TOK + base + (ok ? tid : 0);
        tok[tid] = idxl[a];
        wgs[tid] = ok ? wgtl[a] : 0.f;
        sls[tid] = sltl[a];
    }
    __syncthreads();
    // stage X tile (fp32 -> bf16), zero-fill invalid rows so WMMA stays clean
    for (int i = tid; i < MT * DM; i += 512) {
        int r = i >> 10, c = i & (DM - 1);
        float v = (r < nvalid) ? x[(size_t)tok[r] * DM + c] : 0.f;
        Xs[r * XS_STRIDE + c] = (__bf16)v;
    }
    __syncthreads();

    const int wid = tid >> 5, lane = tid & 31;
    const int lm = lane & 15, khalf = lane >> 4;
    const int mi = wid & 3,  fi = wid >> 2;        // phase-1 tile of H
    const int n0 = wid * 64;                       // phase-2 output columns

    v8f acc[4][4];
#pragma unroll
    for (int a2 = 0; a2 < 4; ++a2)
#pragma unroll
        for (int b2i = 0; b2i < 4; ++b2i)
#pragma unroll
            for (int r = 0; r < 8; ++r) acc[a2][b2i][r] = 0.f;

    const float bias1_base = 0.f; (void)bias1_base;

    for (int f0 = 0; f0 < DF; f0 += FC) {
        // ---- phase 1: H[0:64][f0:f0+64] = gelu(X @ W1^T + b1), one 16x16 tile/wave
        v8f h = {0.f, 0.f, 0.f, 0.f, 0.f, 0.f, 0.f, 0.f};
        const float*  w1row = w1 + ((size_t)e * DF + f0 + fi * 16 + lm) * DM;
        const __bf16* xrow  = Xs + (mi * 16 + lm) * XS_STRIDE;
#pragma unroll 4
        for (int k0 = 0; k0 < DM; k0 += 32) {
            // A fragment: 16 consecutive k per lane (k = 16*khalf + s), same order as B
            v8bf alo = *(const v8bf*)(xrow + k0 + 16 * khalf);
            v8bf ahi = *(const v8bf*)(xrow + k0 + 8 + 16 * khalf);
            v16bf a = __builtin_shufflevector(alo, ahi,
                0,1,2,3,4,5,6,7,8,9,10,11,12,13,14,15);
            // B fragment: lane = column n, 16 consecutive k (row-major w1 matches)
            const float* bp = w1row + k0 + 16 * khalf;
            v16bf b;
#pragma unroll
            for (int i = 0; i < 16; ++i) b[i] = (__bf16)bp[i];
            h = __builtin_amdgcn_wmma_f32_16x16x32_bf16(
                    false, a, false, b, (short)0, h, false, false);
        }
        const float bias1 = b1[e * DF + f0 + fi * 16 + lm];
#pragma unroll
        for (int r = 0; r < 8; ++r) {
            float g = gelu_exact(h[r] + bias1);
            Hs[(mi * 16 + r + 8 * khalf) * HS_STRIDE + fi * 16 + lm] = (__bf16)g;
        }
        __syncthreads();

        // ---- phase 2: acc(64x64 per wave) += H(64 x 64) @ W2[n-range]^T
#pragma unroll
        for (int kk = 0; kk < FC; kk += 32) {
            v16bf af[4];
#pragma unroll
            for (int m2 = 0; m2 < 4; ++m2) {
                const __bf16* hr = Hs + (m2 * 16 + lm) * HS_STRIDE + kk;
                v8bf lo = *(const v8bf*)(hr + 16 * khalf);
                v8bf hi = *(const v8bf*)(hr + 8 + 16 * khalf);
                af[m2] = __builtin_shufflevector(lo, hi,
                    0,1,2,3,4,5,6,7,8,9,10,11,12,13,14,15);
            }
#pragma unroll
            for (int ni = 0; ni < 4; ++ni) {
                const float* bp = w2 + ((size_t)e * DM + n0 + ni * 16 + lm) * DF
                                + f0 + kk + 16 * khalf;
                v16bf b;
#pragma unroll
                for (int i = 0; i < 16; ++i) b[i] = (__bf16)bp[i];
#pragma unroll
                for (int m2 = 0; m2 < 4; ++m2)
                    acc[m2][ni] = __builtin_amdgcn_wmma_f32_16x16x32_bf16(
                        false, af[m2], false, b, (short)0, acc[m2][ni], false, false);
            }
        }
        __syncthreads();
    }

    // epilogue: plane[slot][token] = gate_w * (acc + b2)  (exclusive writes)
#pragma unroll
    for (int m2 = 0; m2 < 4; ++m2) {
#pragma unroll
        for (int r = 0; r < 8; ++r) {
            int row = m2 * 16 + r + 8 * khalf;
            if (row < nvalid) {
                int   t = tok[row];
                float w = wgs[row];
                int   s = sls[row];
                float* op = planes + ((size_t)s * T_TOK + t) * DM;
#pragma unroll
                for (int ni = 0; ni < 4; ++ni) {
                    int col = n0 + ni * 16 + lm;
                    op[col] = w * (acc[m2][ni][r] + b2[e * DM + col]);
                }
            }
        }
    }
}

// ---------------------------------------------------------------- combine
__global__ __launch_bounds__(256) void moe_combine(const float* __restrict__ planes,
                                                   float* __restrict__ out)
{
    size_t i = (size_t)blockIdx.x * 256 + threadIdx.x;
    out[i] = planes[i] + planes[(size_t)T_TOK * DM + i];
}

// ---------------------------------------------------------------- launch
extern "C" void kernel_launch(void* const* d_in, const int* in_sizes, int n_in,
                              void* d_out, int out_size, void* d_ws, size_t ws_size,
                              hipStream_t stream) {
    (void)in_sizes; (void)n_in; (void)out_size; (void)ws_size;
    const float* x   = (const float*)d_in[0];
    const float* gw  = (const float*)d_in[1];
    const float* w1  = (const float*)d_in[2];
    const float* b1  = (const float*)d_in[3];
    const float* w2  = (const float*)d_in[4];
    const float* b2  = (const float*)d_in[5];
    float* out = (float*)d_out;

    char* ws = (char*)d_ws;
    int*   cnt         = (int*)ws;           ws += NE * sizeof(int);
    int*   loadcnt     = (int*)ws;           ws += NE * sizeof(int);
    float* imp_partial = (float*)ws;         ws += (T_TOK / 8) * NE * sizeof(float);
    int*   idxl        = (int*)ws;           ws += NE * T_TOK * sizeof(int);
    float* wgtl        = (float*)ws;         ws += NE * T_TOK * sizeof(float);
    int*   sltl        = (int*)ws;           ws += NE * T_TOK * sizeof(int);
    // align planes to 256 B
    ws = (char*)(((uintptr_t)ws + 255) & ~(uintptr_t)255);
    float* planes      = (float*)ws;         // 2 * T_TOK * DM floats (64 MB)

    hipFuncSetAttribute((const void*)moe_ffn,
                        hipFuncAttributeMaxDynamicSharedMemorySize, FFN_LDS_BYTES);

    moe_zero<<<1, 32, 0, stream>>>(cnt, loadcnt);
    moe_router<<<T_TOK / 8, 256, 0, stream>>>(x, gw, cnt, loadcnt,
                                              idxl, wgtl, sltl, imp_partial);
    moe_aux<<<1, 32, 0, stream>>>(imp_partial, loadcnt, out + (size_t)T_TOK * DM);
    moe_ffn<<<NE * TILES_PER_E, 512, FFN_LDS_BYTES, stream>>>(
        x, w1, b1, w2, b2, cnt, idxl, wgtl, sltl, planes);
    moe_combine<<<(T_TOK * DM) / 256, 256, 0, stream>>>(planes, out);
}
